// FrangiNet_53480932770154
// MI455X (gfx1250) — compile-verified, matrix operand order checked by reading, not run on hardware
//
#include <hip/hip_runtime.h>

typedef __attribute__((ext_vector_type(16))) _Float16 v16h;
typedef __attribute__((ext_vector_type(8)))  _Float16 v8h;
typedef __attribute__((ext_vector_type(8)))  float    v8f;

#define HH 768
#define NB 8
#define NCH 9
#define KPAD 264   // 256 K-halves + 8 pad: 528B rows keep 16B align, stagger banks

union V16 { v16h v; v8h h[2]; };

// ---------------------------------------------------------------- prep
__global__ void prep_kernel(const float* __restrict__ img,
                            float* __restrict__ x1, float* __restrict__ x2,
                            float* __restrict__ x3) {
    int idx = blockIdx.x * blockDim.x + threadIdx.x;
    int total = NB * 192 * 192;
    if (idx >= total) return;
    int x = idx % 192;
    int y = (idx / 192) % 192;
    int b = idx / (192 * 192);
    float s[4][4];
    for (int dy = 0; dy < 4; ++dy)
        for (int dx = 0; dx < 4; ++dx) {
            int gy = y * 4 + dy, gx = x * 4 + dx;
            float v = img[((size_t)b * HH + gy) * HH + gx] * 255.0f;
            s[dy][dx] = v;
            x1[((size_t)b * HH + gy) * HH + gx] = v;
        }
    float m3 = 0.0f;
    for (int dy = 0; dy < 2; ++dy)
        for (int dx = 0; dx < 2; ++dx) {
            float m = 0.25f * (s[2 * dy][2 * dx] + s[2 * dy][2 * dx + 1] +
                               s[2 * dy + 1][2 * dx] + s[2 * dy + 1][2 * dx + 1]);
            x2[((size_t)b * 384 + y * 2 + dy) * 384 + x * 2 + dx] = m;
            m3 += 0.25f * m;
        }
    x3[((size_t)b * 192 + y) * 192 + x] = m3;
}

// --------------------------------------------- hessian conv + frangi + upsample
__global__ void hess_frangi_kernel(const float* __restrict__ src, int W,
                                   const float* __restrict__ wfil, int ksz, int hw,
                                   float* __restrict__ outs, int ch, int S,
                                   const float* __restrict__ betas) {
    __shared__ float tile[34 * 34];     // max halo tile: 16 + 2*9 = 34
    __shared__ float wl[3 * 19 * 19];   // max filter: 3 x 19 x 19
    int tx = threadIdx.x, ty = threadIdx.y;
    int tid = ty * 16 + tx;
    int tw = 16 + 2 * hw;
    int ox0 = blockIdx.x * 16, oy0 = blockIdx.y * 16;
    int b = blockIdx.z;
    const float* sp = src + (size_t)b * W * W;

    for (int i = tid; i < tw * tw; i += 256) {
        int lx = i % tw, ly = i / tw;
        int gx = ox0 - hw + lx, gy = oy0 - hw + ly;
        float v = 0.0f;
        if (gx >= 0 && gx < W && gy >= 0 && gy < W) v = sp[(size_t)gy * W + gx];
        tile[i] = v;
    }
    int kk = ksz * ksz;
    for (int i = tid; i < 3 * kk; i += 256) wl[i] = wfil[i];
    __syncthreads();

    float dxx = 0.f, dxy = 0.f, dyy = 0.f;
    for (int ky = 0; ky < ksz; ++ky) {
        const float* trow = &tile[(ty + ky) * tw + tx];
        const float* w0 = &wl[ky * ksz];
        for (int kx = 0; kx < ksz; ++kx) {
            float v = trow[kx];
            dxx = fmaf(v, w0[kx], dxx);
            dxy = fmaf(v, w0[kk + kx], dxy);
            dyy = fmaf(v, w0[2 * kk + kx], dyy);
        }
    }
    // Frangi 2D vesselness
    float b0 = betas[0], b1 = betas[1];
    float half_tr = 0.5f * (dxx + dyy);
    float disc = sqrtf(0.25f * (dxx - dyy) * (dxx - dyy) + dxy * dxy);
    float la = half_tr - disc, lb = half_tr + disc;
    float e1, e2;
    if (fabsf(lb) < fabsf(la)) { e1 = lb; e2 = la; } else { e1 = la; e2 = lb; }
    float e1s = e1 * e1, e2s = e2 * e2;
    float Rb2 = e1s / (e2s + 1e-6f);
    float S2 = e1s + e2s;
    float blob = expf(-Rb2 / (2.0f * b0 * b0));
    float background = 1.0f - expf(-S2 / (2.0f * b1 * b1));
    float vessel = blob * background;
    if (e2 < 0.0f) vessel = 0.0f;          // IS_DARK
    if (vessel != vessel) vessel = 0.0f;    // NaN -> 0

    int oy = (oy0 + ty) * S, ox = (ox0 + tx) * S;
    for (int sy = 0; sy < S; ++sy)
        for (int sx = 0; sx < S; ++sx)
            outs[(((size_t)b * NCH + ch) * HH + oy + sy) * HH + ox + sx] = vessel;
}

// ---------------------------------------------------------------- max over 9 ch
__global__ void max_kernel(const float* __restrict__ outs, float* __restrict__ maxo) {
    size_t idx = (size_t)blockIdx.x * blockDim.x + threadIdx.x;
    size_t total = (size_t)NB * HH * HH;
    if (idx >= total) return;
    size_t b = idx / ((size_t)HH * HH);
    size_t p = idx % ((size_t)HH * HH);
    float mx = 0.0f;
    for (int c = 0; c < NCH; ++c)
        mx = fmaxf(mx, outs[((size_t)b * NCH + c) * HH * HH + p]);
    maxo[idx] = mx;
}

// -------------------------------- 9->9 5x5 conv + BN as implicit GEMM via WMMA
// Per wave: M = 16 contiguous pixels of one row, N = 16 (9 real out channels),
// K = 9*25 = 225 padded to 256 -> 8 x v_wmma_f32_16x16x32_f16.
// A and B are pre-shaped in LDS as [row][256 f16] im2col tiles so each fragment
// is exactly two contiguous 16B runs -> 2 x ds_load_b128 per operand per WMMA.
__global__ void __launch_bounds__(128)
conv1_wmma_kernel(const float* __restrict__ outs,
                  const float* __restrict__ w_oc1,
                  const float* __restrict__ gamma,
                  const float* __restrict__ beta,
                  const float* __restrict__ mean,
                  const float* __restrict__ var,
                  float* __restrict__ ybuf) {
    __shared__ __align__(16) _Float16 Abuf[4][16 * KPAD]; // per-wave im2col A
    __shared__ __align__(16) _Float16 Bbuf[16 * KPAD];    // shared weight tile
    __shared__ float patch[4][900];                       // 9ch x 5 x 20 (f32)

    int tid = threadIdx.x;              // 0..127, 4 waves
    int lane = tid & 31;
    int wv = tid >> 5;
    int m = lane & 15;                  // pixel (A) / out-channel (B,C,D)
    int h = lane >> 4;                  // lane half

    // ---- Stage B (decode-free): B_lds[n][k] = w_oc1[n*225 + k], zero padded
    for (int i = tid; i < 16 * 256; i += 128) {
        int n = i >> 8, k = i & 255;
        float v = (n < NCH && k < 225) ? w_oc1[n * 225 + k] : 0.0f;
        Bbuf[n * KPAD + k] = (_Float16)v;
    }

    int wid = blockIdx.x * 4 + wv;      // one wave = one (b,row,xtile)
    int xt = wid % 48;
    int row = (wid / 48) % HH;
    int b = wid / (48 * HH);

    // ---- Stage this wave's input patch (zero-padded halo): 9ch x 5 x 20
    float* pp = patch[wv];
    for (int i = lane; i < 900; i += 32) {
        int c = i / 100, r = i % 100;
        int py = r / 20, px = r % 20;
        int gy = row - 2 + py, gx = xt * 16 - 2 + px;
        float v = 0.0f;
        if (gy >= 0 && gy < HH && gx >= 0 && gx < HH)
            v = outs[(((size_t)b * NCH + c) * HH + gy) * HH + gx];
        pp[i] = v;
    }
    __syncthreads();

    // ---- Build im2col A tile, division-free: A[m][c*25+dy*5+dx] = patch[c][dy][m+dx]
    _Float16* am = &Abuf[wv][m * KPAD];
    #pragma unroll 1
    for (int c = 0; c < NCH; ++c) {
        #pragma unroll 1
        for (int dy = 0; dy < 5; ++dy) {
            const float* pr = &pp[c * 100 + dy * 20];
            int k0 = c * 25 + dy * 5;
            for (int d3 = 0; d3 < 3; ++d3) {
                int dx = 2 * d3 + h;                 // lane halves split dx
                if (dx < 5) am[k0 + dx] = (_Float16)pr[m + dx];
            }
        }
    }
    // zero-pad K in [225,256)
    for (int i = lane; i < 16 * 32; i += 32) {
        int mm = i >> 5, kz = 225 + (i & 31);
        if (kz < 256) Abuf[wv][mm * KPAD + kz] = (_Float16)0.0f;
    }
    __syncthreads();

    // ---- Hot loop: 4 x ds_load_b128 + 1 x v_wmma per chunk
    v8f acc = {};
    const _Float16* arow = &Abuf[wv][m * KPAD];
    const _Float16* brow = &Bbuf[m * KPAD];
    #pragma unroll
    for (int chunk = 0; chunk < 8; ++chunk) {
        int kb = chunk * 32 + 8 * h;    // fragment: K = kb..kb+7, kb+16..kb+23
        V16 ua, ub;
        ua.h[0] = *(const v8h*)&arow[kb];
        ua.h[1] = *(const v8h*)&arow[kb + 16];
        ub.h[0] = *(const v8h*)&brow[kb];
        ub.h[1] = *(const v8h*)&brow[kb + 16];
        acc = __builtin_amdgcn_wmma_f32_16x16x32_f16(
            /*neg_a=*/false, ua.v, /*neg_b=*/false, ub.v,
            /*c_mod=*/(short)0, acc, /*reuse_a=*/false, /*reuse_b=*/false);
    }

    // ---- Fused BN + store: lane holds N = m, rows M = r + 8*h
    int n = m;
    if (n < NCH) {
        float inv = gamma[n] * rsqrtf(var[n] + 1e-5f);
        float shift = beta[n] - mean[n] * inv;
        #pragma unroll
        for (int r = 0; r < 8; ++r) {
            int mp = 8 * h + r;
            ybuf[(((size_t)b * NCH + n) * HH + row) * HH + xt * 16 + mp] =
                acc[r] * inv + shift;
        }
    }
}

// ---------------------------------------------------------------- 9->1 3x3 + max
__global__ void conv2_add_kernel(const float* __restrict__ ybuf,
                                 const float* __restrict__ w_oc2,
                                 const float* __restrict__ maxo,
                                 float* __restrict__ out) {
    __shared__ float t2[NCH][18 * 18];  // 16x16 tile + 1-halo per channel
    __shared__ float w2[NCH * 9];
    int tx = threadIdx.x, ty = threadIdx.y;
    int tid = ty * 16 + tx;
    int ox0 = blockIdx.x * 16, oy0 = blockIdx.y * 16;
    int b = blockIdx.z;

    for (int c = 0; c < NCH; ++c) {
        const float* yc = &ybuf[((size_t)b * NCH + c) * HH * HH];
        for (int i = tid; i < 18 * 18; i += 256) {
            int ly = i / 18, lx = i % 18;
            int gy = oy0 - 1 + ly, gx = ox0 - 1 + lx;
            float v = 0.0f;
            if (gy >= 0 && gy < HH && gx >= 0 && gx < HH) v = yc[(size_t)gy * HH + gx];
            t2[c][i] = v;
        }
    }
    if (tid < NCH * 9) w2[tid] = w_oc2[tid];
    __syncthreads();

    int oy = oy0 + ty, ox = ox0 + tx;
    size_t pidx = ((size_t)b * HH + oy) * HH + ox;
    float accv = maxo[pidx];
    #pragma unroll
    for (int c = 0; c < NCH; ++c)
        #pragma unroll
        for (int ky = 0; ky < 3; ++ky)
            #pragma unroll
            for (int kx = 0; kx < 3; ++kx)
                accv = fmaf(t2[c][(ty + ky) * 18 + tx + kx], w2[c * 9 + ky * 3 + kx], accv);
    out[pidx] = accv;
}

// ---------------------------------------------------------------- launch
extern "C" void kernel_launch(void* const* d_in, const int* in_sizes, int n_in,
                              void* d_out, int out_size, void* d_ws, size_t ws_size,
                              hipStream_t stream) {
    (void)in_sizes; (void)n_in; (void)out_size; (void)ws_size;
    const float* img   = (const float*)d_in[0];
    const float* betas = (const float*)d_in[1];
    const float* w_s[3] = {(const float*)d_in[2], (const float*)d_in[3], (const float*)d_in[4]};
    const float* w_oc1 = (const float*)d_in[5];
    const float* w_oc2 = (const float*)d_in[6];
    const float* gamma = (const float*)d_in[7];
    const float* beta  = (const float*)d_in[8];
    const float* mean  = (const float*)d_in[9];
    const float* var   = (const float*)d_in[10];
    float* out = (float*)d_out;

    float* ws = (float*)d_ws;
    float* x1   = ws;                                   // 8*768*768
    float* x2   = x1 + (size_t)NB * HH * HH;            // 8*384*384
    float* x3   = x2 + (size_t)NB * 384 * 384;          // 8*192*192
    float* outs = x3 + (size_t)NB * 192 * 192;          // 8*9*768*768
    float* maxo = outs + (size_t)NB * NCH * HH * HH;    // 8*768*768
    float* ybuf = maxo + (size_t)NB * HH * HH;          // 8*9*768*768

    {
        int total = NB * 192 * 192;
        prep_kernel<<<(total + 255) / 256, 256, 0, stream>>>(img, x1, x2, x3);
    }

    const float* srcs[3] = {x1, x2, x3};
    const int Wv[3] = {768, 384, 192};
    const int Sv[3] = {1, 2, 4};
    const int Kv[3] = {7, 13, 19};
    for (int s = 0; s < 3; ++s) {
        for (int sc = 0; sc < 3; ++sc) {
            dim3 grid(Wv[sc] / 16, Wv[sc] / 16, NB);
            hess_frangi_kernel<<<grid, dim3(16, 16), 0, stream>>>(
                srcs[sc], Wv[sc], w_s[s], Kv[s], Kv[s] / 2,
                outs, s * 3 + sc, Sv[sc], betas);
        }
    }
    {
        size_t total = (size_t)NB * HH * HH;
        max_kernel<<<(unsigned)((total + 255) / 256), 256, 0, stream>>>(outs, maxo);
    }
    {
        int blocks = NB * HH * 48 / 4;   // 4 waves per 128-thread block
        conv1_wmma_kernel<<<blocks, 128, 0, stream>>>(outs, w_oc1, gamma, beta,
                                                      mean, var, ybuf);
    }
    {
        dim3 grid(HH / 16, HH / 16, NB);
        conv2_add_kernel<<<grid, dim3(16, 16), 0, stream>>>(ybuf, w_oc2, maxo, out);
    }
}
